// TridentMOELayer_21053929685426
// MI455X (gfx1250) — matrix operand order repaired and checked by reading, not compiled, not run on hardware
//
#include <hip/hip_runtime.h>

// ---- CDNA5 (gfx1250) vector types for WMMA ----
typedef __attribute__((ext_vector_type(16))) _Float16 v16h;
typedef __attribute__((ext_vector_type(8)))  _Float16 v8h;
typedef __attribute__((ext_vector_type(4)))  _Float16 v4h;
typedef __attribute__((ext_vector_type(8)))  float    v8f;

#define B_TOTAL 8192
#define IN_FEAT 4096
#define NEXP    64
#define DEX     128
#define ICHUNK  32
#define BT      32     // batch rows per block
#define EPB     8      // experts per block

// ---- Kernel 1: ternarize routing matrix (32x64) into workspace ----
__global__ void trident_ternary_k(const float* __restrict__ rm,
                                  const float* __restrict__ noise,
                                  float* __restrict__ tern) {
    int i = blockIdx.x * blockDim.x + threadIdx.x;
    if (i < ICHUNK * NEXP) {
        float wn = rm[i] + 0.1f * noise[i];
        tern[i] = (wn < -0.5f) ? -1.0f : ((wn > 0.5f) ? 1.0f : 0.0f);
    }
}

// ---- Kernel 2: fused routing-reduction + per-expert GEMM via f16 WMMA ----
__global__ __launch_bounds__(256) void trident_moe_k(
    const float* __restrict__ x,     // (B, 4096)
    const float* __restrict__ We,    // (64, 128, 128) row-major [e][n][j]
    const float* __restrict__ tern,  // (32, 64)
    float* __restrict__ out) {       // (B, 8192)
    __shared__ __align__(16) _Float16 sWe[DEX * DEX];  // 32 KB, [n][j]
    __shared__ __align__(16) _Float16 sXr[BT * DEX];   //  8 KB, [r][j]

    const int tid  = threadIdx.x;
    const int lane = tid & 31;
    const int wave = tid >> 5;        // 0..7  -> N-tile index
    const int l15  = lane & 15;
    const int lhi  = lane >> 4;

    const int b0 = blockIdx.x * BT;
    const int e0 = blockIdx.y * EPB;

    // stage-1 assignment: j = column 0..127, rows interleaved by 2
    const int j     = tid & 127;
    const int rbase = tid >> 7;       // 0 or 1

    for (int ee = 0; ee < EPB; ++ee) {
        const int e = e0 + ee;

        // ---- stage 1: x_routed tile (BT x 128), f32 accumulation ----
        float acc[BT / 2];
#pragma unroll
        for (int r = 0; r < BT / 2; ++r) acc[r] = 0.0f;

        for (int i = 0; i < ICHUNK; ++i) {
            const float t = tern[i * NEXP + e];           // uniform address
            // force a scalar (whole-wave) skip of zero routing weights
            if (__builtin_amdgcn_readfirstlane(__float_as_int(t)) != 0) {
                const float* xp = x + (size_t)b0 * IN_FEAT + (size_t)i * DEX + j;
#pragma unroll
                for (int r = 0; r < BT / 2; ++r)
                    acc[r] = fmaf(t, xp[(size_t)(2 * r + rbase) * IN_FEAT], acc[r]);
            }
        }

        __syncthreads();  // previous iteration's WMMA LDS reads are done

#pragma unroll
        for (int r = 0; r < BT / 2; ++r)
            sXr[(2 * r + rbase) * DEX + j] = (_Float16)acc[r];

        // ---- stage We[e] -> LDS as f16 (vectorized: float4 -> 4x f16) ----
        const float4* wp4  = (const float4*)(We + (size_t)e * DEX * DEX);
        v4h*          sWe4 = (v4h*)sWe;
#pragma unroll 4
        for (int k = tid; k < DEX * DEX / 4; k += 256) {
            float4 wv = wp4[k];
            v4h hv;
            hv[0] = (_Float16)wv.x; hv[1] = (_Float16)wv.y;
            hv[2] = (_Float16)wv.z; hv[3] = (_Float16)wv.w;
            sWe4[k] = hv;
        }

        __syncthreads();

        // ---- WMMA: out_tile(BT x 128) = xr(BT x 128) * We[e]^T ----
        // 2 M-tiles x 8 N-tiles; wave w owns N-tile w; K = 128 in 4 steps of 32
#pragma unroll
        for (int m = 0; m < BT / 16; ++m) {
            v8f c = {};
#pragma unroll
            for (int ks = 0; ks < 4; ++ks) {
                union { v16h v; v8h h[2]; } a, b;
                // A 16x32 f16 layout: lane l15=M row; 16 contiguous K halves
                // starting at K = ks*32 + lhi*16
                const _Float16* ap = &sXr[(m * 16 + l15) * DEX + ks * 32];
                a.h[0] = *(const v8h*)(ap + lhi * 16);
                a.h[1] = *(const v8h*)(ap + lhi * 16 + 8);
                // B 32x16 f16 layout: lane l15=N col; 16 contiguous K halves
                // starting at K = ks*32 + lhi*16, from row N of We (K = j)
                const _Float16* bp = &sWe[(wave * 16 + l15) * DEX + ks * 32 + lhi * 16];
                b.h[0] = *(const v8h*)(bp);
                b.h[1] = *(const v8h*)(bp + 8);
                c = __builtin_amdgcn_wmma_f32_16x16x32_f16(
                        /*neg_a=*/false, a.v, /*neg_b=*/false, b.v,
                        /*c_mod=*/(short)0, c, /*reuse_a=*/false, /*reuse_b=*/false);
            }
            // C/D f32 layout: VGPR v -> M = v + lhi*8, N = l15
            float* op = out + (size_t)(b0 + m * 16 + lhi * 8) * (NEXP * DEX)
                            + (size_t)e * DEX + wave * 16 + l15;
#pragma unroll
            for (int v = 0; v < 8; ++v)
                op[(size_t)v * (NEXP * DEX)] = c[v];
        }
    }
}

extern "C" void kernel_launch(void* const* d_in, const int* in_sizes, int n_in,
                              void* d_out, int out_size, void* d_ws, size_t ws_size,
                              hipStream_t stream) {
    const float* x     = (const float*)d_in[0];  // (8192, 4096)
    const float* rm    = (const float*)d_in[1];  // (32, 64)
    const float* We    = (const float*)d_in[2];  // (64, 128, 128)
    const float* noise = (const float*)d_in[3];  // (32, 64)
    float*       out   = (float*)d_out;          // (8192, 8192)
    float*       tern  = (float*)d_ws;           // 8 KB scratch

    trident_ternary_k<<<dim3((ICHUNK * NEXP + 255) / 256), dim3(256), 0, stream>>>(
        rm, noise, tern);

    trident_moe_k<<<dim3(B_TOTAL / BT, NEXP / EPB), dim3(256), 0, stream>>>(
        x, We, tern, out);
}
